// Rethinking_24739011625431
// MI455X (gfx1250) — compile-verified, matrix operand order checked by reading, not run on hardware
//
#include <hip/hip_runtime.h>
#include <hip/hip_bf16.h>
#include <math.h>

// Problem constants (from reference): B=32, T=1024, VOCAB=512, RANK=256
#define BATCH 32
#define SEQ   1024
#define VOC   512
#define RNK   256
#define N_QKV (3 * RNK)   // 768

typedef __attribute__((ext_vector_type(16))) __bf16 v16bf;
typedef __attribute__((ext_vector_type(8)))  float  v8f;
typedef __attribute__((ext_vector_type(4)))  unsigned int u32x4;
typedef __attribute__((ext_vector_type(8)))  unsigned int u32x8;

union BFrag {
    v16bf v;
    unsigned int u[8];
};

// round-to-nearest-even f32 -> bf16 bits
__device__ __forceinline__ unsigned short f2bf(float f) {
    unsigned int u = __float_as_uint(f);
    u += 0x7fffu + ((u >> 16) & 1u);
    return (unsigned short)(u >> 16);
}

// Load a 16x32 bf16 fragment (A-operand layout, ISA 7.12.2) from LDS.
// `base` points at element [row 0][k 0]; rows have stride `ldr` elements.
// Same layout serves B when the B tile is staged [n][k] in LDS.
__device__ __forceinline__ v16bf frag_ld(const unsigned short* base, int ldr) {
    const int lane = threadIdx.x & 31;
    const int r = lane & 15;
    const int h = lane >> 4;
    const unsigned short* rp = base + r * ldr + h * 8;
    BFrag f;
#pragma unroll
    for (int g = 0; g < 8; ++g) {
        const int kk = 2 * (g & 3) + 16 * (g >> 2);
        f.u[g] = *(const unsigned int*)(rp + kk);
    }
    return f.v;
}

__device__ __forceinline__ v8f wmma_bf16(v16bf a, v16bf b, v8f c) {
    return __builtin_amdgcn_wmma_f32_16x16x32_bf16(
        false, a, false, b, (short)0, c, false, false);
}

// ---------------------------------------------------------------------------
// Tensor Data Mover: DMA a 2D bf16 tile (tile_x elems wide, tile_y rows,
// row stride `stride` elems) from global memory into LDS (ISA ch.8 D#).
// Issue from ONE wave; pair with s_wait_tensorcnt + workgroup barrier.
// ---------------------------------------------------------------------------
__device__ __forceinline__ void tdm_load_2d_bf16(
    const unsigned short* lds_dst, const unsigned short* gsrc,
    unsigned tile_x, unsigned tile_y, unsigned long long stride,
    unsigned tensor_x, unsigned tensor_y) {
    const unsigned long long ga = (unsigned long long)(uintptr_t)gsrc;
    const unsigned lds_off = (unsigned)(unsigned long long)(uintptr_t)lds_dst;
    u32x4 g0;
    g0[0] = 1u;                                  // count=1 valid user D#
    g0[1] = lds_off;                             // lds_addr (bytes)
    g0[2] = (unsigned)ga;                        // global_addr[31:0]
    g0[3] = (unsigned)((ga >> 32) & 0x1FFFFFFull) | (2u << 30); // addr hi | type=2
    u32x8 g1;
    g1[0] = 1u << 16;                            // data_size = 2 bytes
    g1[1] = (tensor_x & 0xFFFFu) << 16;          // tensor_dim0[15:0]
    g1[2] = (tensor_x >> 16) | ((tensor_y & 0xFFFFu) << 16); // dim0 hi | dim1 lo
    g1[3] = (tensor_y >> 16) | (tile_x << 16);   // dim1 hi | tile_dim0
    g1[4] = tile_y;                              // tile_dim1 (tile_dim2 = 0)
    g1[5] = (unsigned)stride;                    // tensor_dim0_stride[31:0]
    g1[6] = (unsigned)(stride >> 32) & 0xFFFFu;  // stride hi (dim1_stride = 0)
    g1[7] = 0u;
    asm volatile("tensor_load_to_lds %0, %1" :: "s"(g0), "s"(g1) : "memory");
}

// ---------------------------------------------------------------------------
// Kernel 1: qkv = x @ W_attn + b_attn     (f32 in, bf16 out, bf16 WMMA)
// M=32768 K=512 N=768.  Block tile 128x64, 8 waves of 32x32, K-step 32.
// ---------------------------------------------------------------------------
#define BM 128
#define BN 64
#define BK 32

__global__ __launch_bounds__(256) void gemm_qkv(
    const float* __restrict__ X, const float* __restrict__ W,
    const float* __restrict__ bias, unsigned short* __restrict__ out) {
    __shared__ __align__(16) unsigned short sA[BM * BK]; // [m][k]
    __shared__ __align__(16) unsigned short sB[BN * BK]; // [n][k]

    const int bm = blockIdx.x * BM;
    const int bn = blockIdx.y * BN;
    const int tid = threadIdx.x;
    const int wave = tid >> 5, lane = tid & 31;
    const int wm = (wave & 3) * 32;
    const int wn = (wave >> 2) * 32;
    const int r = lane & 15, h = lane >> 4;

    v8f acc[2][2];
    const v8f vzero = {0.f, 0.f, 0.f, 0.f, 0.f, 0.f, 0.f, 0.f};
#pragma unroll
    for (int i = 0; i < 2; ++i)
#pragma unroll
        for (int j = 0; j < 2; ++j) acc[i][j] = vzero;

    for (int k0 = 0; k0 < VOC; k0 += BK) {
        // Stage A: b128 loads (float4), convert, b64 LDS stores
#pragma unroll
        for (int q = tid; q < BM * BK / 4; q += 256) {
            const int m = q >> 3;          // 8 quads per 32-elem row
            const int kk = (q & 7) * 4;
            const float4 xv = *(const float4*)&X[(size_t)(bm + m) * VOC + k0 + kk];
            *(ushort4*)&sA[m * BK + kk] =
                make_ushort4(f2bf(xv.x), f2bf(xv.y), f2bf(xv.z), f2bf(xv.w));
        }
        // Stage B transposed to [n][k]: b128 global loads, b16 scattered stores
#pragma unroll
        for (int q = tid; q < BN * BK / 4; q += 256) {
            const int kk = q >> 4;         // 16 quads per 64-elem n-row
            const int n = (q & 15) * 4;
            const float4 wv = *(const float4*)&W[(size_t)(k0 + kk) * N_QKV + bn + n];
            sB[(n + 0) * BK + kk] = f2bf(wv.x);
            sB[(n + 1) * BK + kk] = f2bf(wv.y);
            sB[(n + 2) * BK + kk] = f2bf(wv.z);
            sB[(n + 3) * BK + kk] = f2bf(wv.w);
        }
        if (k0 + BK < VOC) // prefetch next K-slab of A (global_prefetch_b8)
            __builtin_prefetch(&X[(size_t)(bm + (tid & 127)) * VOC + k0 + BK], 0, 1);
        __syncthreads();

        v16bf b0 = frag_ld(&sB[(wn + 0) * BK], BK);
        v16bf b1 = frag_ld(&sB[(wn + 16) * BK], BK);
#pragma unroll
        for (int im = 0; im < 2; ++im) {
            v16bf a = frag_ld(&sA[(wm + im * 16) * BK], BK);
            acc[im][0] = wmma_bf16(a, b0, acc[im][0]);
            acc[im][1] = wmma_bf16(a, b1, acc[im][1]);
        }
        __syncthreads();
    }

#pragma unroll
    for (int im = 0; im < 2; ++im)
#pragma unroll
        for (int in = 0; in < 2; ++in)
#pragma unroll
            for (int v = 0; v < 8; ++v) {
                const int row = bm + wm + im * 16 + v + 8 * h;
                const int col = bn + wn + in * 16 + r;
                out[(size_t)row * N_QKV + col] = f2bf(acc[im][in][v] + bias[col]);
            }
}

// ---------------------------------------------------------------------------
// Kernel 2: causal flash attention over bf16 q/k/v (slices of qkv).
// One block = 16 query rows, 4 waves; K tile arrives via the TDM, V tile is
// staged transposed by the VALUs, P round-trips LDS into A-fragment layout.
// ---------------------------------------------------------------------------
#define KT 32  // key tile

__global__ __launch_bounds__(128) void attn(
    const unsigned short* __restrict__ qkv, unsigned short* __restrict__ Y) {
    __shared__ __align__(16) unsigned short sQ[16 * RNK];   // [row][d]
    __shared__ __align__(16) unsigned short sK[KT * RNK];   // [key][d]  (TDM)
    __shared__ __align__(16) unsigned short sVt[RNK * KT];  // [d][key]
    __shared__ __align__(16) unsigned short sP[4][16 * KT]; // per-wave P

    const int qbase = blockIdx.x * 16;
    const int batch = blockIdx.y;
    const size_t tok0 = (size_t)batch * SEQ + qbase;
    const int tid = threadIdx.x, wave = tid >> 5, lane = tid & 31;
    const int r = lane & 15, h = lane >> 4;

    // Stage Q tile (bf16 copy, b64 chunks)
    for (int q = tid; q < 16 * RNK / 4; q += 128) {
        const int row = q >> 6, d = (q & 63) * 4;
        *(ushort4*)&sQ[row * RNK + d] =
            *(const ushort4*)&qkv[(tok0 + row) * N_QKV + d];
    }
    __syncthreads();

    const v8f vzero = {0.f, 0.f, 0.f, 0.f, 0.f, 0.f, 0.f, 0.f};
    v8f o[4];
#pragma unroll
    for (int t = 0; t < 4; ++t) o[t] = vzero;
    float m_i[8], l_i[8];
#pragma unroll
    for (int v = 0; v < 8; ++v) { m_i[v] = -1e30f; l_i[v] = 0.f; }

    const int kend = qbase + 16; // causal: keys <= last query row
    for (int ks = 0; ks < kend; ks += KT) {
        // K tile via Tensor Data Mover: 32 rows x 256 elems, row stride 768
        if (wave == 0)
            tdm_load_2d_bf16(sK, &qkv[((size_t)batch * SEQ + ks) * N_QKV + RNK],
                             RNK, KT, N_QKV, RNK, (unsigned)(BATCH * SEQ));
        // V tile transposed by VALUs (TDM cannot transpose)
        for (int q = tid; q < KT * RNK / 4; q += 128) {
            const int key = q >> 6, d = (q & 63) * 4;
            const size_t tok = (size_t)batch * SEQ + ks + key;
            const ushort4 vv = *(const ushort4*)&qkv[tok * N_QKV + 2 * RNK + d];
            sVt[(d + 0) * KT + key] = vv.x;
            sVt[(d + 1) * KT + key] = vv.y;
            sVt[(d + 2) * KT + key] = vv.z;
            sVt[(d + 3) * KT + key] = vv.w;
        }
        if (wave == 0) __builtin_amdgcn_s_wait_tensorcnt(0);
        __syncthreads();

        // S = q @ k^T over d=256
        v8f s0 = vzero, s1 = vzero;
#pragma unroll
        for (int dk = 0; dk < RNK; dk += 32) {
            v16bf a  = frag_ld(&sQ[dk], RNK);
            v16bf b0 = frag_ld(&sK[0 * RNK + dk], RNK);
            v16bf b1 = frag_ld(&sK[16 * RNK + dk], RNK);
            s0 = wmma_bf16(a, b0, s0);
            s1 = wmma_bf16(a, b1, s1);
        }

        // online softmax; C-layout: element v -> row qbase+v+8h, col ks(+16)+r
        const float scale = 0.0625f; // 1/sqrt(256)
#pragma unroll
        for (int v = 0; v < 8; ++v) {
            const int rowg = qbase + v + 8 * h;
            float e0 = s0[v] * scale;
            float e1 = s1[v] * scale;
            if (ks + r      > rowg) e0 = -1e30f;
            if (ks + 16 + r > rowg) e1 = -1e30f;
            float mx = fmaxf(e0, e1);
#pragma unroll
            for (int off = 8; off >= 1; off >>= 1)
                mx = fmaxf(mx, __shfl_xor(mx, off, 32)); // stays in half-wave
            const float mnew = fmaxf(m_i[v], mx);
            const float alpha = __expf(m_i[v] - mnew);
            e0 = __expf(e0 - mnew);
            e1 = __expf(e1 - mnew);
            float rs = e0 + e1;
#pragma unroll
            for (int off = 8; off >= 1; off >>= 1)
                rs += __shfl_xor(rs, off, 32);
            l_i[v] = l_i[v] * alpha + rs;
            m_i[v] = mnew;
#pragma unroll
            for (int t = 0; t < 4; ++t) o[t][v] *= alpha;
            sP[wave][(v + 8 * h) * KT + r]      = f2bf(e0);
            sP[wave][(v + 8 * h) * KT + 16 + r] = f2bf(e1);
        }
        __syncthreads();

        // O += P @ V : wave owns out dims [wave*64, wave*64+64)
        v16bf pa = frag_ld(&sP[wave][0], KT);
#pragma unroll
        for (int t = 0; t < 4; ++t) {
            v16bf vb = frag_ld(&sVt[(wave * 64 + t * 16) * KT], KT);
            o[t] = wmma_bf16(pa, vb, o[t]);
        }
        __syncthreads();
    }

#pragma unroll
    for (int t = 0; t < 4; ++t)
#pragma unroll
        for (int v = 0; v < 8; ++v) {
            const size_t tok = tok0 + v + 8 * h;
            Y[tok * RNK + wave * 64 + t * 16 + r] = f2bf(o[t][v] / l_i[v]);
        }
}

// ---------------------------------------------------------------------------
// Kernel 3: out = gelu_exact(y @ W_proj + b_proj)   (bf16 A via TDM, f32 out)
// M=32768 K=256 N=512.
// ---------------------------------------------------------------------------
__global__ __launch_bounds__(256) void gemm_proj(
    const unsigned short* __restrict__ Ybf, const float* __restrict__ W,
    const float* __restrict__ bias, float* __restrict__ out) {
    __shared__ __align__(16) unsigned short sA[BM * BK];
    __shared__ __align__(16) unsigned short sB[BN * BK];

    const int bm = blockIdx.x * BM;
    const int bn = blockIdx.y * BN;
    const int tid = threadIdx.x;
    const int wave = tid >> 5, lane = tid & 31;
    const int wm = (wave & 3) * 32;
    const int wn = (wave >> 2) * 32;
    const int r = lane & 15, h = lane >> 4;

    v8f acc[2][2];
    const v8f vzero = {0.f, 0.f, 0.f, 0.f, 0.f, 0.f, 0.f, 0.f};
#pragma unroll
    for (int i = 0; i < 2; ++i)
#pragma unroll
        for (int j = 0; j < 2; ++j) acc[i][j] = vzero;

    for (int k0 = 0; k0 < RNK; k0 += BK) {
        // A tile is a pure bf16 2D copy -> Tensor Data Mover (wave 0)
        if (wave == 0)
            tdm_load_2d_bf16(sA, &Ybf[(size_t)bm * RNK + k0],
                             BK, BM, RNK, RNK, (unsigned)(BATCH * SEQ));
        // B tile: f32 -> bf16 conversion on the VALUs, overlapped with TDM
#pragma unroll
        for (int q = tid; q < BN * BK / 4; q += 256) {
            const int kk = q >> 4;
            const int n = (q & 15) * 4;
            const float4 wv = *(const float4*)&W[(size_t)(k0 + kk) * VOC + bn + n];
            sB[(n + 0) * BK + kk] = f2bf(wv.x);
            sB[(n + 1) * BK + kk] = f2bf(wv.y);
            sB[(n + 2) * BK + kk] = f2bf(wv.z);
            sB[(n + 3) * BK + kk] = f2bf(wv.w);
        }
        if (k0 + BK < RNK)
            __builtin_prefetch(&Ybf[(size_t)(bm + (tid & 127)) * RNK + k0 + BK], 0, 1);
        if (wave == 0) __builtin_amdgcn_s_wait_tensorcnt(0);
        __syncthreads();

        v16bf b0 = frag_ld(&sB[(wn + 0) * BK], BK);
        v16bf b1 = frag_ld(&sB[(wn + 16) * BK], BK);
#pragma unroll
        for (int im = 0; im < 2; ++im) {
            v16bf a = frag_ld(&sA[(wm + im * 16) * BK], BK);
            acc[im][0] = wmma_bf16(a, b0, acc[im][0]);
            acc[im][1] = wmma_bf16(a, b1, acc[im][1]);
        }
        __syncthreads();
    }

#pragma unroll
    for (int im = 0; im < 2; ++im)
#pragma unroll
        for (int in = 0; in < 2; ++in)
#pragma unroll
            for (int v = 0; v < 8; ++v) {
                const int row = bm + wm + im * 16 + v + 8 * h;
                const int col = bn + wn + in * 16 + r;
                const float x = acc[im][in][v] + bias[col];
                out[(size_t)row * VOC + col] =
                    0.5f * x * (1.0f + erff(x * 0.70710678118654752f));
            }
}

// ---------------------------------------------------------------------------
extern "C" void kernel_launch(void* const* d_in, const int* in_sizes, int n_in,
                              void* d_out, int out_size, void* d_ws, size_t ws_size,
                              hipStream_t stream) {
    (void)in_sizes; (void)n_in; (void)out_size; (void)ws_size;
    const float* x  = (const float*)d_in[0];
    const float* Wa = (const float*)d_in[1];
    const float* ba = (const float*)d_in[2];
    const float* Wp = (const float*)d_in[3];
    const float* bp = (const float*)d_in[4];
    float* out = (float*)d_out;

    // Workspace (bf16): qkv [32768 x 768] then y [32768 x 256]  (~64 MB)
    unsigned short* qkv = (unsigned short*)d_ws;
    unsigned short* y   = qkv + (size_t)BATCH * SEQ * N_QKV;

    dim3 g1((BATCH * SEQ) / BM, N_QKV / BN);
    gemm_qkv<<<g1, 256, 0, stream>>>(x, Wa, ba, qkv);

    dim3 g2(SEQ / 16, BATCH);
    attn<<<g2, 128, 0, stream>>>(qkv, y);

    dim3 g3((BATCH * SEQ) / BM, VOC / BN);
    gemm_proj<<<g3, 256, 0, stream>>>(y, Wp, bp, out);
}
